// DVMixtureSynthesizers_3418793968313
// MI455X (gfx1250) — compile-verified, matrix operand order checked by reading, not run on hardware
//
#include <hip/hip_runtime.h>
#include <math.h>

// ---------------------------------------------------------------------------
// Mixture synthesizer attention for MI455X (gfx1250), bf16 WMMA everywhere.
//   B=16, S=2048, D=1024, H=64
//   1) hq    = relu(query @ w1 + b1)               f32 in  -> f32  [B*S, H]
//   2) attn  = hq @ w2 + b2                        f32 in  -> f32  [B*S, S]
//   3) qp16  = query @ wq + bq                     f32 in  -> bf16 [B*S, D]
//   4) kp16  = key   @ wk + bk                     f32 in  -> bf16 [B*S, D]
//   5) attn[b] += qp16[b] @ kp16[b]^T              bf16 in -> f32  (batched)
//   6) softmax rows of attn (f32) + bf16 copy attn16 (reuses qp16/kp16 space)
//   7) out[b] = attn16[b] @ value[b]               bf16/f32 -> f32 (batched)
// All matmuls: v_wmma_f32_16x16x32_bf16, 128x128x32 block tile, 8 waves,
// register-prefetch + double-buffered LDS (one barrier per K-step).
// ---------------------------------------------------------------------------

#define BM 128
#define BN 128
#define BK 32
#define KSTR 40   // BK + 8 halfword pad: 80B row stride, keeps 16B alignment

typedef __attribute__((ext_vector_type(16))) __bf16 v16bf;
typedef __attribute__((ext_vector_type(8)))  float  v8f;

struct __align__(16) U4 { unsigned int x, y, z, w; };
union Frag { v16bf v; U4 q[2]; };

enum { F_RELU = 1, F_ADDC = 2 };

__device__ __forceinline__ unsigned short f2bf(float f) {
  union { float f; unsigned int u; } c; c.f = f;          // round-to-nearest-even
  return (unsigned short)((c.u + 0x7FFFu + ((c.u >> 16) & 1u)) >> 16);
}
__device__ __forceinline__ unsigned int pk2bf(float x, float y) {
  return (unsigned int)f2bf(x) | ((unsigned int)f2bf(y) << 16);
}

// ---------------------------------------------------------------------------
// Templated GEMM. A: [M,K] (f32 or bf16, row-major, lda).
// B: BT16 ? bf16 [N,K] row-major (ldb)  :  f32 [K,N] row-major (ldb).
// C: OUT16 ? bf16 : f32, [M,N] (ldc). M always a multiple of BM (asserted by
// launches). GUARDN enables ragged-N handling (only the tiny H=64 MLP).
// ---------------------------------------------------------------------------
template <bool ABF16, bool BT16, bool GUARDN, bool OUT16>
__global__ __launch_bounds__(256) void gemm_wmma(
    const void* __restrict__ Aq, long long sA, int lda,
    const void* __restrict__ Bq, long long sB, int ldb,
    const float* __restrict__ bias,
    void* __restrict__ Cq, long long sC, int ldc,
    int M, int N, int K, int flags) {
  __shared__ unsigned short As[2][BM * KSTR];
  __shared__ unsigned short Bs[2][BN * KSTR];

  const float*          Af = (const float*)Aq          + (ABF16 ? 0 : (long long)blockIdx.z * sA);
  const unsigned short* Ab = (const unsigned short*)Aq + (ABF16 ? (long long)blockIdx.z * sA : 0);
  const float*          Bf = (const float*)Bq          + (BT16 ? 0 : (long long)blockIdx.z * sB);
  const unsigned short* Bb = (const unsigned short*)Bq + (BT16 ? (long long)blockIdx.z * sB : 0);

  const int bm0 = blockIdx.y * BM;
  const int bn0 = blockIdx.x * BN;
  const int tid  = threadIdx.x;
  const int lane = tid & 31;          // wave32
  const int wave = tid >> 5;          // 8 waves: 2 (M) x 4 (N)
  const int wm   = (wave & 1) * 64;
  const int wn   = (wave >> 1) * 32;
  const int half = lane >> 4;
  const int l16  = lane & 15;
  const int kb   = half * 8;

  const v8f vzero = {0.f, 0.f, 0.f, 0.f, 0.f, 0.f, 0.f, 0.f};
  v8f acc[4][2];
  #pragma unroll
  for (int i = 0; i < 4; ++i)
    #pragma unroll
    for (int j = 0; j < 2; ++j) acc[i][j] = vzero;

  // prefetch registers
  float4 pa[4]; U4 pa16[2];
  float4 pb[4]; U4 pb16[2];

  auto loadA = [&](int k0) {
    if constexpr (ABF16) {
      #pragma unroll
      for (int it = 0; it < 2; ++it) {
        int lin = tid + 256 * it;          // 512 x b128 covers 128x32 bf16
        int r = lin >> 2, q = lin & 3;
        pa16[it] = *(const U4*)(Ab + (long long)(bm0 + r) * lda + (k0 + q * 8));
      }
    } else {
      #pragma unroll
      for (int it = 0; it < 4; ++it) {
        int r = (tid >> 3) + 32 * it, c = (tid & 7) * 4;
        pa[it] = *(const float4*)(Af + (long long)(bm0 + r) * lda + (k0 + c));
      }
    }
  };
  auto storeA = [&](unsigned short* buf) {
    if constexpr (ABF16) {
      #pragma unroll
      for (int it = 0; it < 2; ++it) {
        int lin = tid + 256 * it;
        int r = lin >> 2, q = lin & 3;
        *(U4*)&buf[r * KSTR + q * 8] = pa16[it];
      }
    } else {
      #pragma unroll
      for (int it = 0; it < 4; ++it) {
        int r = (tid >> 3) + 32 * it, c = (tid & 7) * 4;
        unsigned int* d = (unsigned int*)&buf[r * KSTR + c];
        d[0] = pk2bf(pa[it].x, pa[it].y);
        d[1] = pk2bf(pa[it].z, pa[it].w);
      }
    }
  };
  auto loadB = [&](int k0) {
    if constexpr (BT16) {                  // bf16 [N,K]: natural copy
      #pragma unroll
      for (int it = 0; it < 2; ++it) {
        int lin = tid + 256 * it;
        int n = lin >> 2, q = lin & 3;
        pb16[it] = *(const U4*)(Bb + (long long)(bn0 + n) * ldb + (k0 + q * 8));
      }
    } else {                               // f32 [K,N]: coalesced along n
      #pragma unroll
      for (int it = 0; it < 4; ++it) {
        int k = (tid >> 5) + 8 * it, n4 = (tid & 31) * 4;
        if (!GUARDN || bn0 + n4 < N)
          pb[it] = *(const float4*)(Bf + (long long)(k0 + k) * ldb + (bn0 + n4));
        else
          pb[it] = make_float4(0.f, 0.f, 0.f, 0.f);
      }
    }
  };
  auto storeB = [&](unsigned short* buf) {
    if constexpr (BT16) {
      #pragma unroll
      for (int it = 0; it < 2; ++it) {
        int lin = tid + 256 * it;
        int n = lin >> 2, q = lin & 3;
        *(U4*)&buf[n * KSTR + q * 8] = pb16[it];
      }
    } else {                               // transposed scatter: Bs[n][k]
      #pragma unroll
      for (int it = 0; it < 4; ++it) {
        int k = (tid >> 5) + 8 * it, n4 = (tid & 31) * 4;
        buf[(n4 + 0) * KSTR + k] = f2bf(pb[it].x);
        buf[(n4 + 1) * KSTR + k] = f2bf(pb[it].y);
        buf[(n4 + 2) * KSTR + k] = f2bf(pb[it].z);
        buf[(n4 + 3) * KSTR + k] = f2bf(pb[it].w);
      }
    }
  };

  // --- software pipeline: prefetch k+1 during WMMAs on k, ping-pong LDS ---
  loadA(0); loadB(0);
  storeA(As[0]); storeB(Bs[0]);
  __syncthreads();

  int cur = 0;
  for (int k0 = 0; k0 < K; k0 += BK) {
    const bool has_next = (k0 + BK) < K;
    if (has_next) { loadA(k0 + BK); loadB(k0 + BK); }

    const unsigned short* Ac = As[cur];
    const unsigned short* Bc = Bs[cur];
    Frag fa[4], fb[2];
    #pragma unroll
    for (int i = 0; i < 4; ++i) {
      const unsigned short* p = &Ac[(wm + i * 16 + l16) * KSTR + kb];
      fa[i].q[0] = *(const U4*)p;
      fa[i].q[1] = *(const U4*)(p + 16);
    }
    #pragma unroll
    for (int j = 0; j < 2; ++j) {
      const unsigned short* p = &Bc[(wn + j * 16 + l16) * KSTR + kb];
      fb[j].q[0] = *(const U4*)p;
      fb[j].q[1] = *(const U4*)(p + 16);
    }
    #pragma unroll
    for (int i = 0; i < 4; ++i)
      #pragma unroll
      for (int j = 0; j < 2; ++j)
        acc[i][j] = __builtin_amdgcn_wmma_f32_16x16x32_bf16(
            false, fa[i].v, false, fb[j].v, (short)0, acc[i][j], false, false);

    if (has_next) { storeA(As[cur ^ 1]); storeB(Bs[cur ^ 1]); }
    __syncthreads();
    cur ^= 1;
  }

  // --- epilogue: lane = column l16, VGPR r = row r + 8*half ---
  #pragma unroll
  for (int j = 0; j < 2; ++j) {
    int col = bn0 + wn + j * 16 + l16;
    if (GUARDN && col >= N) continue;
    float bv = bias ? bias[col] : 0.f;
    #pragma unroll
    for (int i = 0; i < 4; ++i) {
      #pragma unroll
      for (int r = 0; r < 8; ++r) {
        int row = bm0 + wm + i * 16 + half * 8 + r;
        float v = acc[i][j][r] + bv;
        long long off = (long long)row * ldc + col;
        if constexpr (OUT16) {
          unsigned short* Co = (unsigned short*)Cq + (long long)blockIdx.z * sC;
          Co[off] = f2bf(v);
        } else {
          float* Co = (float*)Cq + (long long)blockIdx.z * sC;
          if (flags & F_ADDC) v += Co[off];
          if (flags & F_RELU) v = fmaxf(v, 0.f);
          Co[off] = v;
        }
      }
    }
  }
}

// Row softmax in place (f32) + bf16 copy for the attn @ V GEMM.
__global__ __launch_bounds__(256) void softmax_rows(float* __restrict__ attn,
                                                    unsigned short* __restrict__ attn16,
                                                    int n) {
  float* row = attn + (long long)blockIdx.x * n;
  unsigned short* row16 = attn16 + (long long)blockIdx.x * n;
  const int tid = threadIdx.x;
  __shared__ float redm[8];
  __shared__ float reds[8];

  float m = -INFINITY;
  for (int c = tid; c < n; c += 256) m = fmaxf(m, row[c]);
  #pragma unroll
  for (int o = 16; o > 0; o >>= 1) m = fmaxf(m, __shfl_xor(m, o, 32));
  if ((tid & 31) == 0) redm[tid >> 5] = m;
  __syncthreads();
  float bm = redm[0];
  #pragma unroll
  for (int i = 1; i < 8; ++i) bm = fmaxf(bm, redm[i]);

  float s = 0.f;
  for (int c = tid; c < n; c += 256) {
    float e = __expf(row[c] - bm);
    row[c] = e;
    s += e;
  }
  #pragma unroll
  for (int o = 16; o > 0; o >>= 1) s += __shfl_xor(s, o, 32);
  if ((tid & 31) == 0) reds[tid >> 5] = s;
  __syncthreads();
  float bs = 0.f;
  #pragma unroll
  for (int i = 0; i < 8; ++i) bs += reds[i];
  float inv = 1.0f / bs;
  for (int c = tid; c < n; c += 256) {
    float p = row[c] * inv;
    row[c] = p;
    row16[c] = f2bf(p);
  }
}

extern "C" void kernel_launch(void* const* d_in, const int* in_sizes, int n_in,
                              void* d_out, int out_size, void* d_ws, size_t ws_size,
                              hipStream_t stream) {
  (void)in_sizes; (void)n_in; (void)out_size; (void)ws_size;
  const int Bn = 16, S = 2048, D = 1024, H = 64;
  const long long BSD = (long long)Bn * S * D;
  const int MBS = Bn * S;  // 32768 rows

  const float* query = (const float*)d_in[0];
  const float* key   = (const float*)d_in[1];
  const float* value = (const float*)d_in[2];
  const float* w1    = (const float*)d_in[3];
  const float* b1    = (const float*)d_in[4];
  const float* w2    = (const float*)d_in[5];
  const float* b2    = (const float*)d_in[6];
  const float* wq    = (const float*)d_in[7];
  const float* bq    = (const float*)d_in[8];
  const float* wk    = (const float*)d_in[9];
  const float* bk    = (const float*)d_in[10];

  float* out  = (float*)d_out;                  // [B,S,D]
  float* attn = out + BSD;                      // [B,S,S] second output (f32)

  // Workspace (142 MB): qp16/kp16 are dead after step 5 -> attn16 reuses them.
  unsigned short* qp16   = (unsigned short*)d_ws;         // [B*S, D] bf16
  unsigned short* kp16   = qp16 + BSD;                    // [B*S, D] bf16
  unsigned short* attn16 = qp16;                          // [B*S, S] bf16 (reuse)
  float*          hq     = (float*)(kp16 + BSD);          // [B*S, H] f32

  dim3 blk(256);

  // 1) hq = relu(query @ w1 + b1)        M=32768 N=64   K=1024  (ragged N)
  gemm_wmma<false, false, true, false>
      <<<dim3(1, MBS / BM, 1), blk, 0, stream>>>(
      query, 0, D, w1, 0, H, b1, hq, 0, H, MBS, H, D, F_RELU);
  // 2) attn = hq @ w2 + b2               M=32768 N=2048 K=64
  gemm_wmma<false, false, false, false>
      <<<dim3(S / BN, MBS / BM, 1), blk, 0, stream>>>(
      hq, 0, H, w2, 0, S, b2, attn, 0, S, MBS, S, H, 0);
  // 3) qp16 = bf16(query @ wq + bq)      M=32768 N=1024 K=1024
  gemm_wmma<false, false, false, true>
      <<<dim3(D / BN, MBS / BM, 1), blk, 0, stream>>>(
      query, 0, D, wq, 0, D, bq, qp16, 0, D, MBS, D, D, 0);
  // 4) kp16 = bf16(key @ wk + bk)
  gemm_wmma<false, false, false, true>
      <<<dim3(D / BN, MBS / BM, 1), blk, 0, stream>>>(
      key, 0, D, wk, 0, D, bk, kp16, 0, D, MBS, D, D, 0);
  // 5) attn[b] += qp16[b] @ kp16[b]^T    batched: M=N=2048 K=1024, bf16 x bf16
  gemm_wmma<true, true, false, false>
      <<<dim3(S / BN, S / BM, Bn), blk, 0, stream>>>(
      qp16, (long long)S * D, D, kp16, (long long)S * D, D, nullptr,
      attn, (long long)S * S, S, S, S, D, F_ADDC);
  // 6) softmax rows (f32 in place) + bf16 copy
  softmax_rows<<<dim3(MBS), blk, 0, stream>>>(attn, attn16, S);
  // 7) out[b] = attn16[b] @ value[b]     batched: M=2048 N=1024 K=2048
  gemm_wmma<true, false, false, false>
      <<<dim3(D / BN, S / BM, Bn), blk, 0, stream>>>(
      attn16, (long long)S * S, S, value, (long long)S * D, D, nullptr,
      out, (long long)S * D, D, S, D, S, 0);
}
